// AttentionConditionGenerator_40991167873578
// MI455X (gfx1250) — matrix operand order repaired and checked by reading, not compiled, not run on hardware
//
#include <hip/hip_runtime.h>
#include <hip/hip_bf16.h>
#include <math.h>

// ---------------------------------------------------------------------------
// Fused attention-condition block for gfx1250 (MI455X).
// B=65536, D=256, H=8, DH=32, FFN=1024.
// All GEMMs via v_wmma_f32_16x16x32_bf16 (bf16 in, f32 accumulate).
// ---------------------------------------------------------------------------

typedef __attribute__((ext_vector_type(16))) __bf16     v16bf;
typedef __attribute__((ext_vector_type(8)))  float      v8f;
typedef __attribute__((ext_vector_type(4)))  unsigned   v4u;

#define DIM      256
#define HEADS    8
#define FFNDIM   1024
#define MROWS    64          // rows per workgroup
#define NTHREADS 256         // 8 wave32
#define PITCH    264         // bf16 elements per LDS tile row (bank-conflict pad)
#define PITCHB   (PITCH*2)   // 528 bytes
#define XPITCH   260         // fp32 elements per LDS row
#define XPITCHB  (XPITCH*4)  // 1040 bytes

// LDS layout (phase-aliased)
#define OFF_A0   0            // qn        | later: x (fp32) spans A0+A1
#define OFF_A1   33792        // kn0
#define OFF_A2   67584        // kn1
#define OFF_T0   101376       // q   -> ctx        | later: result (fp32) spans T0+T1
#define OFF_T1   135168       // k0  -> xn
#define OFF_T2   168960       // k1  -> ffn hidden chunk
#define OFF_SC   202752       // attention a0 probs: 64*8 fp32
#define SMEM_BYTES 204800
#define OFF_X    OFF_A0
#define OFF_R    OFF_T0

// ---- bf16 helpers (round-to-nearest-even) ---------------------------------
__device__ __forceinline__ unsigned f2bf1(float f) {
  unsigned u = __float_as_uint(f);
  return (u + 0x7FFFu + ((u >> 16) & 1u)) >> 16;
}
__device__ __forceinline__ unsigned packbf(float lo, float hi) {
  return f2bf1(lo) | (f2bf1(hi) << 16);
}
__device__ __forceinline__ float bflo(unsigned u) { return __uint_as_float(u << 16); }
__device__ __forceinline__ float bfhi(unsigned u) { return __uint_as_float(u & 0xFFFF0000u); }

union FragU { v4u q[2]; v16bf v; };

// A-fragment (16x32 bf16) from row-major LDS tile, per ISA 7.12.2 layout:
// lane<16 : row = mt*16+lane, K = kb*32 + {0..7, 16..23}
// lane>=16: row = mt*16+(lane-16), K = kb*32 + {8..15, 24..31}
__device__ __forceinline__ v16bf load_a_frag(const char* smem, int bufOff, int mt,
                                             int kb, int lane) {
  int row = mt * 16 + (lane & 15);
  int kh  = lane >> 4;
  const char* p = smem + bufOff + row * PITCHB + (kb * 32 + kh * 8) * 2;
  FragU f;
  f.q[0] = *(const v4u*)(p);
  f.q[1] = *(const v4u*)(p + 32);
  return f.v;
}

// B-fragment (32x16 bf16) pre-swizzled in workspace: 1024B per fragment,
// lane-major, 32B per lane.
__device__ __forceinline__ v16bf load_b_frag(const v4u* wf, int frag, int lane) {
  const v4u* p = wf + frag * 64 + lane * 2;
  FragU f;
  f.q[0] = p[0];
  f.q[1] = p[1];
  return f.v;
}

__device__ __forceinline__ v8f wmma_bf16(v16bf a, v16bf b, v8f c) {
  return __builtin_amdgcn_wmma_f32_16x16x32_bf16(false, a, false, b, (short)0, c,
                                                 false, false);
}

// ---------------------------------------------------------------------------
// Prep kernel: fp32 row-major W (N x K) -> bf16 B-fragment layout.
// dst dword t: d = t&7, lane = (t>>3)&31, frag = t>>8 = nt*KB + kb.
// lane n-part selects output column n = nt*16 + (lane&15);
// K = kb*32 + (lane>=16 ? 8 : 0) + 2*d + (d>=4 ? 8 : 0)  (pairs packed low/high).
// ---------------------------------------------------------------------------
__global__ void prep_wfrag(const float* __restrict__ W, unsigned* __restrict__ dst,
                           int K, int total) {
  int t = blockIdx.x * blockDim.x + threadIdx.x;
  if (t >= total) return;
  int d    = t & 7;
  int lane = (t >> 3) & 31;
  int kbnt = t >> 8;
  int KB   = K >> 5;
  int kb   = kbnt % KB;
  int nt   = kbnt / KB;
  int n    = nt * 16 + (lane & 15);
  int k    = kb * 32 + (lane >> 4) * 8 + d * 2 + ((d >> 2) << 3);
  const float* wr = W + (long)n * K + k;
  dst[t] = packbf(wr[0], wr[1]);
}

// ---------------------------------------------------------------------------
// Main fused kernel.
// ---------------------------------------------------------------------------
__global__ __launch_bounds__(NTHREADS)
void fused_attn_ffn_kernel(
    const float* __restrict__ cross, const float* __restrict__ srcp,
    const float* __restrict__ tgtp,
    const float* __restrict__ inb,  const float* __restrict__ outb,
    const float* __restrict__ ln1g, const float* __restrict__ ln1b,
    const float* __restrict__ ln2g, const float* __restrict__ ln2b,
    const float* __restrict__ b1,   const float* __restrict__ b2,
    const v4u* __restrict__ wqf, const v4u* __restrict__ wkf,
    const v4u* __restrict__ wvf, const v4u* __restrict__ wof,
    const v4u* __restrict__ w1f, const v4u* __restrict__ w2f,
    float* __restrict__ out) {
  extern __shared__ char smem[];
  const int tid  = threadIdx.x;
  const int lane = tid & 31;
  const int wave = tid >> 5;
  const long r0  = (long)blockIdx.x * MROWS;

  // ---- P1: LayerNorm(ln1) of cross/source/target -> A0/A1/A2 (bf16) ------
  {
    float gv[8], bv[8];
    *(float4*)(gv)     = *(const float4*)(ln1g + lane * 8);
    *(float4*)(gv + 4) = *(const float4*)(ln1g + lane * 8 + 4);
    *(float4*)(bv)     = *(const float4*)(ln1b + lane * 8);
    *(float4*)(bv + 4) = *(const float4*)(ln1b + lane * 8 + 4);
    for (int i = 0; i < 24; i++) {
      int rt = wave + i * 8;          // 0..191
      int t  = rt >> 6;               // 0:cross 1:source 2:target
      int r  = rt & 63;
      const float* sp = (t == 0) ? cross : ((t == 1) ? srcp : tgtp);
      const float4* rp = (const float4*)(sp + (r0 + r) * DIM + lane * 8);
      float4 x0 = rp[0], x1 = rp[1];
      float v[8] = {x0.x, x0.y, x0.z, x0.w, x1.x, x1.y, x1.z, x1.w};
      float s = 0.f, s2 = 0.f;
      for (int j = 0; j < 8; j++) { s += v[j]; s2 += v[j] * v[j]; }
      for (int o = 16; o > 0; o >>= 1) {
        s += __shfl_xor(s, o);
        s2 += __shfl_xor(s2, o);
      }
      float mu = s * (1.f / 256.f);
      float rs = rsqrtf(s2 * (1.f / 256.f) - mu * mu + 1e-5f);
      v4u pk;
      for (int j = 0; j < 4; j++) {
        float a = (v[2 * j] - mu) * rs * gv[2 * j] + bv[2 * j];
        float b = (v[2 * j + 1] - mu) * rs * gv[2 * j + 1] + bv[2 * j + 1];
        pk[j] = packbf(a, b);
      }
      int bufOff = (t == 0) ? OFF_A0 : ((t == 1) ? OFF_A1 : OFF_A2);
      *(v4u*)(smem + bufOff + r * PITCHB + lane * 16) = pk;
    }
  }
  __syncthreads();

  // ---- P2a: q / k0 / k1 GEMMs (WMMA) -> T0 / T1 / T2 (bf16) --------------
  for (int i = 0; i < 24; i++) {
    int f   = wave + i * 8;           // 0..191
    int nt  = f & 15;
    int mt  = (f >> 4) & 3;
    int mat = f >> 6;                 // 0:q 1:k0 2:k1
    int aOff = (mat == 0) ? OFF_A0 : ((mat == 1) ? OFF_A1 : OFF_A2);
    const v4u* wf = (mat == 0) ? wqf : wkf;
    float bias = inb[((mat == 0) ? 0 : 256) + nt * 16 + (lane & 15)];
    v8f acc = {0.f, 0.f, 0.f, 0.f, 0.f, 0.f, 0.f, 0.f};
    for (int kb = 0; kb < 8; kb++) {
      __builtin_prefetch(wf + (nt * 8 + kb + 1) * 64 + lane * 2, 0, 3);
      v16bf a = load_a_frag(smem, aOff, mt, kb, lane);
      v16bf b = load_b_frag(wf, nt * 8 + kb, lane);
      acc = wmma_bf16(a, b, acc);
    }
    int oOff  = (mat == 0) ? OFF_T0 : ((mat == 1) ? OFF_T1 : OFF_T2);
    int col   = nt * 16 + (lane & 15);
    int rbase = mt * 16 + ((lane >> 4) & 1) * 8;
    for (int ii = 0; ii < 8; ii++)
      *(unsigned short*)(smem + oOff + (rbase + ii) * PITCHB + col * 2) =
          (unsigned short)f2bf1(acc[ii] + bias);
  }
  __syncthreads();

  // ---- P2b: 2-way softmax scores, a0 -> OFF_SC ---------------------------
  for (int j = 0; j < 2; j++) {
    int p = tid + j * 256;            // 0..511 = (row, head)
    int m = p >> 3, h = p & 7;
    const unsigned* q  = (const unsigned*)(smem + OFF_T0 + m * PITCHB + h * 64);
    const unsigned* k0 = (const unsigned*)(smem + OFF_T1 + m * PITCHB + h * 64);
    const unsigned* k1 = (const unsigned*)(smem + OFF_T2 + m * PITCHB + h * 64);
    float s0 = 0.f, s1 = 0.f;
    for (int d = 0; d < 16; d++) {
      unsigned uq = q[d], u0 = k0[d], u1 = k1[d];
      float ql = bflo(uq), qh = bfhi(uq);
      s0 += ql * bflo(u0) + qh * bfhi(u0);
      s1 += ql * bflo(u1) + qh * bfhi(u1);
    }
    float a0 = 1.f / (1.f + __expf((s1 - s0) * 0.17677669529663689f));
    *(float*)(smem + OFF_SC + p * 4) = a0;
  }
  __syncthreads();

  // ---- P2c: ctx = a0*v0 + (1-a0)*v1 + bv, fused on WMMA accs -> T0 -------
  for (int i = 0; i < 8; i++) {
    int f = wave + i * 8;             // 0..63
    int nt = f & 15, mt = f >> 4;
    int h  = nt >> 1;                 // head is constant within a 16-col tile
    float bias = inb[512 + nt * 16 + (lane & 15)];
    v8f a0acc = {0.f, 0.f, 0.f, 0.f, 0.f, 0.f, 0.f, 0.f};
    v8f a1acc = {0.f, 0.f, 0.f, 0.f, 0.f, 0.f, 0.f, 0.f};
    for (int kb = 0; kb < 8; kb++) {
      v16bf b  = load_b_frag(wvf, nt * 8 + kb, lane);
      v16bf a0 = load_a_frag(smem, OFF_A1, mt, kb, lane);
      a0acc = wmma_bf16(a0, b, a0acc);
      v16bf a1 = load_a_frag(smem, OFF_A2, mt, kb, lane);
      a1acc = wmma_bf16(a1, b, a1acc);
    }
    int col   = nt * 16 + (lane & 15);
    int rbase = mt * 16 + ((lane >> 4) & 1) * 8;
    for (int ii = 0; ii < 8; ii++) {
      int row = rbase + ii;
      float p0 = *(const float*)(smem + OFF_SC + (row * 8 + h) * 4);
      float v  = p0 * a0acc[ii] + (1.f - p0) * a1acc[ii] + bias;
      *(unsigned short*)(smem + OFF_T0 + row * PITCHB + col * 2) =
          (unsigned short)f2bf1(v);
    }
  }
  __syncthreads();

  // ---- P3a: out-projection (WMMA) -> attn_out in x buffer (fp32) ---------
  for (int i = 0; i < 8; i++) {
    int f = wave + i * 8;
    int nt = f & 15, mt = f >> 4;
    float bias = outb[nt * 16 + (lane & 15)];
    v8f acc = {0.f, 0.f, 0.f, 0.f, 0.f, 0.f, 0.f, 0.f};
    for (int kb = 0; kb < 8; kb++) {
      v16bf a = load_a_frag(smem, OFF_T0, mt, kb, lane);
      v16bf b = load_b_frag(wof, nt * 8 + kb, lane);
      acc = wmma_bf16(a, b, acc);
    }
    int col   = nt * 16 + (lane & 15);
    int rbase = mt * 16 + ((lane >> 4) & 1) * 8;
    for (int ii = 0; ii < 8; ii++)
      *(float*)(smem + OFF_X + (rbase + ii) * XPITCHB + col * 4) = acc[ii] + bias;
  }
  __syncthreads();

  // ---- P3b: x = cross + attn_out (kept fp32) ; LN2(x) -> xn in T1 --------
  {
    float gv[8], bv[8];
    *(float4*)(gv)     = *(const float4*)(ln2g + lane * 8);
    *(float4*)(gv + 4) = *(const float4*)(ln2g + lane * 8 + 4);
    *(float4*)(bv)     = *(const float4*)(ln2b + lane * 8);
    *(float4*)(bv + 4) = *(const float4*)(ln2b + lane * 8 + 4);
    for (int i = 0; i < 8; i++) {
      int r = wave + i * 8;
      char* xrow = smem + OFF_X + r * XPITCHB;
      float4 a0 = *(float4*)(xrow + lane * 32);
      float4 a1 = *(float4*)(xrow + lane * 32 + 16);
      const float4* cp = (const float4*)(cross + (r0 + r) * DIM + lane * 8);
      float4 c0 = cp[0], c1 = cp[1];
      float v[8] = {c0.x + a0.x, c0.y + a0.y, c0.z + a0.z, c0.w + a0.w,
                    c1.x + a1.x, c1.y + a1.y, c1.z + a1.z, c1.w + a1.w};
      *(float4*)(xrow + lane * 32)      = make_float4(v[0], v[1], v[2], v[3]);
      *(float4*)(xrow + lane * 32 + 16) = make_float4(v[4], v[5], v[6], v[7]);
      float s = 0.f, s2 = 0.f;
      for (int j = 0; j < 8; j++) { s += v[j]; s2 += v[j] * v[j]; }
      for (int o = 16; o > 0; o >>= 1) {
        s += __shfl_xor(s, o);
        s2 += __shfl_xor(s2, o);
      }
      float mu = s * (1.f / 256.f);
      float rs = rsqrtf(s2 * (1.f / 256.f) - mu * mu + 1e-5f);
      v4u pk;
      for (int j = 0; j < 4; j++)
        pk[j] = packbf((v[2 * j] - mu) * rs * gv[2 * j] + bv[2 * j],
                       (v[2 * j + 1] - mu) * rs * gv[2 * j + 1] + bv[2 * j + 1]);
      *(v4u*)(smem + OFF_T1 + r * PITCHB + lane * 16) = pk;
    }
  }
  __syncthreads();

  // ---- P4/P5: FFN in 4 chunks of 256 hidden cols; ffn2 accs in VGPRs -----
  v8f facc[8];
  for (int i = 0; i < 8; i++)
    facc[i] = (v8f){0.f, 0.f, 0.f, 0.f, 0.f, 0.f, 0.f, 0.f};
  for (int c = 0; c < 4; c++) {
    // FFN1: h = gelu(xn @ W1^T + b1) for hidden cols [c*256, c*256+256) -> T2
    for (int i = 0; i < 8; i++) {
      int f = wave + i * 8;
      int nt = f & 15, mt = f >> 4;
      int ntg = c * 16 + nt;
      float bias = b1[ntg * 16 + (lane & 15)];
      v8f acc = {0.f, 0.f, 0.f, 0.f, 0.f, 0.f, 0.f, 0.f};
      for (int kb = 0; kb < 8; kb++) {
        __builtin_prefetch(w1f + (ntg * 8 + kb + 1) * 64 + lane * 2, 0, 3);
        v16bf a = load_a_frag(smem, OFF_T1, mt, kb, lane);
        v16bf b = load_b_frag(w1f, ntg * 8 + kb, lane);
        acc = wmma_bf16(a, b, acc);
      }
      int col   = nt * 16 + (lane & 15);
      int rbase = mt * 16 + ((lane >> 4) & 1) * 8;
      for (int ii = 0; ii < 8; ii++) {
        float x = acc[ii] + bias;
        float g = 0.5f * x * (1.f + erff(x * 0.70710678118654752f));
        *(unsigned short*)(smem + OFF_T2 + (rbase + ii) * PITCHB + col * 2) =
            (unsigned short)f2bf1(g);
      }
    }
    __syncthreads();
    // FFN2 partial: facc += h_chunk @ W2[:, chunk]^T
    for (int i = 0; i < 8; i++) {
      int f = wave * 8 + i;           // fixed per-wave mapping across chunks
      int nt = f & 15, mt = f >> 4;
      for (int kb = 0; kb < 8; kb++) {
        __builtin_prefetch(w2f + (nt * 32 + c * 8 + kb + 1) * 64 + lane * 2, 0, 3);
        v16bf a = load_a_frag(smem, OFF_T2, mt, kb, lane);
        v16bf b = load_b_frag(w2f, nt * 32 + c * 8 + kb, lane);
        facc[i] = wmma_bf16(a, b, facc[i]);
      }
    }
    __syncthreads();
  }

  // ---- P6: result = x + ffn + b2 -> staged fp32 in LDS -------------------
  for (int i = 0; i < 8; i++) {
    int f = wave * 8 + i;
    int nt = f & 15, mt = f >> 4;
    int col = nt * 16 + (lane & 15);
    float bias = b2[col];
    int rbase = mt * 16 + ((lane >> 4) & 1) * 8;
    for (int ii = 0; ii < 8; ii++) {
      int row = rbase + ii;
      float xv = *(const float*)(smem + OFF_X + row * XPITCHB + col * 4);
      *(float*)(smem + OFF_R + row * XPITCHB + col * 4) = xv + facc[i][ii] + bias;
    }
  }
  __syncthreads();

  // ---- P7: coalesced row-major store to global ---------------------------
  for (int i = 0; i < 8; i++) {
    int r = wave + i * 8;
    const float4* rp = (const float4*)(smem + OFF_R + r * XPITCHB + lane * 32);
    float4* op = (float4*)(out + (r0 + r) * DIM + lane * 8);
    op[0] = rp[0];
    op[1] = rp[1];
  }
}

// ---------------------------------------------------------------------------
extern "C" void kernel_launch(void* const* d_in, const int* in_sizes, int n_in,
                              void* d_out, int out_size, void* d_ws, size_t ws_size,
                              hipStream_t stream) {
  const float* cross = (const float*)d_in[0];
  const float* srcp  = (const float*)d_in[1];
  const float* tgtp  = (const float*)d_in[2];
  const float* in_w  = (const float*)d_in[3];   // (768,256)
  const float* in_b  = (const float*)d_in[4];   // (768)
  const float* out_w = (const float*)d_in[5];   // (256,256)
  const float* out_b = (const float*)d_in[6];
  const float* ln1g  = (const float*)d_in[7];
  const float* ln1b  = (const float*)d_in[8];
  const float* ln2g  = (const float*)d_in[9];
  const float* ln2b  = (const float*)d_in[10];
  const float* w1    = (const float*)d_in[11];  // (1024,256)
  const float* b1    = (const float*)d_in[12];
  const float* w2    = (const float*)d_in[13];  // (256,1024)
  const float* b2    = (const float*)d_in[14];

  unsigned* ws  = (unsigned*)d_ws;
  unsigned* wqf = ws;                 // 32768 dwords each for 256x256 matrices
  unsigned* wkf = ws + 32768;
  unsigned* wvf = ws + 65536;
  unsigned* wof = ws + 98304;
  unsigned* w1f = ws + 131072;        // 131072 dwords
  unsigned* w2f = ws + 262144;        // 131072 dwords

  // Weight -> bf16 B-fragment swizzle (deterministic, rerun every call).
  prep_wfrag<<<128, 256, 0, stream>>>(in_w,             wqf, 256,  32768);
  prep_wfrag<<<128, 256, 0, stream>>>(in_w + 256 * 256, wkf, 256,  32768);
  prep_wfrag<<<128, 256, 0, stream>>>(in_w + 512 * 256, wvf, 256,  32768);
  prep_wfrag<<<128, 256, 0, stream>>>(out_w,            wof, 256,  32768);
  prep_wfrag<<<512, 256, 0, stream>>>(w1,               w1f, 256,  131072);
  prep_wfrag<<<512, 256, 0, stream>>>(w2,               w2f, 1024, 131072);

  (void)hipFuncSetAttribute((const void*)fused_attn_ffn_kernel,
                            hipFuncAttributeMaxDynamicSharedMemorySize, SMEM_BYTES);

  fused_attn_ffn_kernel<<<65536 / MROWS, NTHREADS, SMEM_BYTES, stream>>>(
      cross, srcp, tgtp, in_b, out_b, ln1g, ln1b, ln2g, ln2b, b1, b2,
      (const v4u*)wqf, (const v4u*)wkf, (const v4u*)wvf, (const v4u*)wof,
      (const v4u*)w1f, (const v4u*)w2f, (float*)d_out);
}
